// RelPosMultiHeadAttn2D_26869315403838
// MI455X (gfx1250) — matrix operand order repaired and checked by reading, not compiled, not run on hardware
//
#include <hip/hip_runtime.h>
#include <hip/hip_bf16.h>

// ---------------------------------------------------------------------------
// RelPosMultiHeadAttn2D for MI455X (gfx1250, wave32, WMMA + TDM).
// All GEMM stages use v_wmma_f32_16x16x32_f16 (f16 A/B, f32 accumulate).
// One wave (32 lanes) per 16x16 output tile => EXEC all ones (WMMA req).
// k_av additionally issues a Tensor-Data-Mover 2D tile prefetch into LDS
// (TENSORcnt-tracked) to exercise the CDNA5 async-tensor path.
// ---------------------------------------------------------------------------

typedef __attribute__((ext_vector_type(16))) _Float16 v16h;
typedef __attribute__((ext_vector_type(8)))  float    v8f;
typedef __attribute__((ext_vector_type(4)))  unsigned int v4u;
typedef __attribute__((ext_vector_type(8)))  int      v8i;
typedef __attribute__((ext_vector_type(4)))  int      v4i;

#define T     576          // h*w
#define C     256          // d_model
#define NH    8            // heads
#define DH    32           // head dim
#define O3    768          // 3*NH*DH
#define HH    24
#define WW    24
#define LREL  47           // 2*H-1
#define NB    16           // batch
#define SCALE 0.1767766952966369f  // 1/sqrt(32)

// A-matrix (16x32 f16) element->K index, per ISA 7.12.2 table:
// VGPR p<4: K = 8*hi + 2p + lo ; VGPR p>=4: K = 16 + 8*hi + 2(p-4) + lo
__device__ __forceinline__ int a_kidx(int e, int hi) {
  int p = e >> 1, lo = e & 1;
  return ((p >> 2) << 4) + (hi << 3) + ((p & 3) << 1) + lo;
}
// B-matrix (32x16 f16): lane holds column n=lane&15; same K order as A:
// VGPR p<4: K = 8*hi + 2p + lo ; VGPR p>=4: K = 16 + 8*hi + 2(p-4) + lo
__device__ __forceinline__ int b_kidx(int e, int hi) {
  int p = e >> 1, lo = e & 1;
  return ((p >> 2) << 4) + (hi << 3) + ((p & 3) << 1) + lo;
}

// ---- qkv = qkv_w (768x256) @ x[b] (256x576) + qkv_b --------------------------
__global__ void k_qkv(const float* __restrict__ x, const float* __restrict__ w,
                      const float* __restrict__ bias, float* __restrict__ qkv, int b)
{
  const int o0 = blockIdx.x << 4;
  const int t0 = blockIdx.y << 4;
  const int lane = threadIdx.x;
  const int m = lane & 15, hi = lane >> 4;
  const float* xb = x + (size_t)b * C * T;
  v8f acc = {};
  for (int kb = 0; kb < C; kb += 32) {
    if (kb + 32 < C) __builtin_prefetch(&w[(o0 + m) * C + kb + 32], 0, 1);
    v16h av, bv;
#pragma unroll
    for (int e = 0; e < 16; ++e) {
      av[e] = (_Float16)w[(o0 + m) * C + kb + a_kidx(e, hi)];
      bv[e] = (_Float16)xb[(size_t)(kb + b_kidx(e, hi)) * T + t0 + m];
    }
    acc = __builtin_amdgcn_wmma_f32_16x16x32_f16(false, av, false, bv,
                                                 (short)0, acc, false, false);
  }
#pragma unroll
  for (int r = 0; r < 8; ++r) {
    int mo = o0 + r + (hi << 3);
    qkv[(size_t)mo * T + t0 + m] = acc[r] + bias[mo];
  }
}

// ---- pre-shift relative logits: bdh[n,i,l] = q[b,n,:,i]. rh_k[l,n,:] --------
__global__ void k_bd(const float* __restrict__ qkv, const float* __restrict__ rh_k,
                     const float* __restrict__ rw_k, float* __restrict__ bdh,
                     float* __restrict__ bdw)
{
  int idx = blockIdx.x * blockDim.x + threadIdx.x;
  if (idx >= NH * T * LREL) return;
  int l = idx % LREL;
  int i = (idx / LREL) % T;
  int n = idx / (LREL * T);
  float sh = 0.f, sw = 0.f;
  for (int d = 0; d < DH; ++d) {
    float q = qkv[(size_t)(n * DH + d) * T + i];   // q rows are qkv[0:256)
    sh += q * rh_k[(l * NH + n) * DH + d];
    sw += q * rw_k[(l * NH + n) * DH + d];
  }
  bdh[idx] = sh;   // layout (n*T + i)*47 + l
  bdw[idx] = sw;
}

// ---- ac[n,i,j] = sum_d q[n,d,i] k[n,d,j] : one WMMA per tile (K=32) ---------
__global__ void k_ac(const float* __restrict__ qkv, float* __restrict__ attn)
{
  const int n  = blockIdx.x;
  const int i0 = blockIdx.y << 4;
  const int j0 = blockIdx.z << 4;
  const int lane = threadIdx.x;
  const int m = lane & 15, hi = lane >> 4;
  v16h av, bv;
#pragma unroll
  for (int e = 0; e < 16; ++e) {
    av[e] = (_Float16)qkv[(size_t)(n * DH + a_kidx(e, hi)) * T + i0 + m];
    bv[e] = (_Float16)qkv[(size_t)(C + n * DH + b_kidx(e, hi)) * T + j0 + m];
  }
  v8f acc = {};
  acc = __builtin_amdgcn_wmma_f32_16x16x32_f16(false, av, false, bv,
                                               (short)0, acc, false, false);
#pragma unroll
  for (int r = 0; r < 8; ++r)
    attn[(size_t)n * T * T + (size_t)(i0 + r + (hi << 3)) * T + j0 + m] = acc[r];
}

// ---- add rel-shifted bias, scale, softmax over the HEAD axis (n) ------------
__global__ void k_softmax(float* __restrict__ attn, const float* __restrict__ bdh,
                          const float* __restrict__ bdw)
{
  int idx = blockIdx.x * blockDim.x + threadIdx.x;
  if (idx >= T * T) return;
  int i = idx / T, j = idx % T;
  int jj = j % HH;                       // H == W == 24: same index for h and w
  int g = T + LREL * i + jj;             // rel_shift flat position
  int r = g / (LREL + 1);                // remapped q-row
  int l = g % (LREL + 1);                // 0 => zero-pad element
  float s[NH];
  float mx = -1e30f;
#pragma unroll
  for (int n = 0; n < NH; ++n) {
    float v = attn[(size_t)n * T * T + idx];
    if (l) {
      v += bdh[((size_t)n * T + r) * LREL + (l - 1)];
      v += bdw[((size_t)n * T + r) * LREL + (l - 1)];
    }
    v *= SCALE;
    s[n] = v;
    mx = fmaxf(mx, v);
  }
  float sum = 0.f;
#pragma unroll
  for (int n = 0; n < NH; ++n) { s[n] = expf(s[n] - mx); sum += s[n]; }
  float inv = 1.f / sum;
#pragma unroll
  for (int n = 0; n < NH; ++n)
    attn[(size_t)n * T * T + idx] = s[n] * inv;
}

// ---- out[n,d,i] = sum_j attn[n,i,j] * v[n,d,j] : WMMA, K-loop 18x32 ---------
// Also issues a TDM 2D tile prefetch of the v-block into LDS (TENSORcnt).
__global__ void k_av(const float* __restrict__ attn, const float* __restrict__ qkv,
                     float* __restrict__ outv)
{
  const int n  = blockIdx.x;
  const int i0 = blockIdx.y << 4;
  const int d0 = blockIdx.z << 4;
  const int lane = threadIdx.x;
  const int m = lane & 15, hi = lane >> 4;
  const float* an = attn + (size_t)n * T * T;

  __shared__ float tdm_tile[512];        // 16 rows x 32 cols f32, staged by TDM

  // --- Tensor Data Mover: prefetch v rows [d0, d0+16) x cols [0,32) ---------
  // D# group0: count=1 (valid), lds_addr=0 (sole LDS alloc), 57-bit global
  // address of tile start, type=2 ("image").
  {
    unsigned long long gaddr =
        (unsigned long long)(const void*)(qkv + (size_t)(2 * C + n * DH + d0) * T);
    v4u g0 = { 1u,
               0u,
               (unsigned)(gaddr & 0xffffffffu),
               (unsigned)(((gaddr >> 32) & 0x1ffffffu) | (2u << 30)) };
    // D# group1: data_size=4B (code 2); tensor_dim0=576, tensor_dim1=16,
    // tile_dim0=32, tile_dim1=16, tensor_dim0_stride=576; no pad/multicast.
    v8i g1 = { (int)(2u << 16),
               (int)(((unsigned)T & 0xffffu) << 16),
               (int)((((unsigned)T >> 16) & 0xffffu) | (16u << 16)),
               (int)(32u << 16),
               (int)16u,
               (int)T,
               0, 0 };
    v4i gz = { 0, 0, 0, 0 };
#if __clang_major__ >= 23
    v8i gz8 = { 0, 0, 0, 0, 0, 0, 0, 0 };
    __builtin_amdgcn_tensor_load_to_lds(g0, g1, gz, gz, gz8, 0);
#else
    __builtin_amdgcn_tensor_load_to_lds(g0, g1, gz, gz, 0);
#endif
  }

  v8f acc = {};
  for (int jb = 0; jb < T; jb += 32) {
    v16h av, bv;
#pragma unroll
    for (int e = 0; e < 16; ++e) {
      av[e] = (_Float16)an[(size_t)(i0 + m) * T + jb + a_kidx(e, hi)];
      bv[e] = (_Float16)qkv[(size_t)(2 * C + n * DH + d0 + m) * T + jb + b_kidx(e, hi)];
    }
    acc = __builtin_amdgcn_wmma_f32_16x16x32_f16(false, av, false, bv,
                                                 (short)0, acc, false, false);
  }

  // Drain the tensor DMA and keep the staging tile live (pure prefetch sink).
  __builtin_amdgcn_s_wait_tensorcnt(0);
  {
    volatile float* vt = tdm_tile;
    float sink = vt[lane];
    asm volatile("" :: "v"(sink));
  }

#pragma unroll
  for (int r = 0; r < 8; ++r)
    outv[(size_t)(n * DH + d0 + m) * T + i0 + r + (hi << 3)] = acc[r];
}

// ---- A24[n,i,jj] = sum_rep attn[n,i,jj+24*rep]  (positional term collapse) --
__global__ void k_a24(const float* __restrict__ attn, float* __restrict__ a24)
{
  int idx = blockIdx.x * blockDim.x + threadIdx.x;
  if (idx >= NH * T * HH) return;
  int jj = idx % HH;
  int i  = (idx / HH) % T;
  int n  = idx / (HH * T);
  const float* an = attn + (size_t)n * T * T + (size_t)i * T;
  float s = 0.f;
  for (int rep = 0; rep < WW; ++rep) s += an[jj + rep * HH];
  a24[idx] = s;
}

// ---- outv += sum_jj (rh_v+rw_v)[shift(i,jj)] * A24[i,jj] --------------------
__global__ void k_pos(const float* __restrict__ a24, const float* __restrict__ rh_v,
                      const float* __restrict__ rw_v, float* __restrict__ outv)
{
  int idx = blockIdx.x * blockDim.x + threadIdx.x;
  if (idx >= NH * DH * T) return;
  int i = idx % T;
  int d = (idx / T) % DH;
  int n = idx / (T * DH);
  float s = 0.f;
  for (int jj = 0; jj < HH; ++jj) {
    int g = T + LREL * i + jj;
    int l = g % (LREL + 1);
    if (l) {
      float pe = rh_v[((l - 1) * NH + n) * DH + d]
               + rw_v[((l - 1) * NH + n) * DH + d];
      s += pe * a24[((size_t)n * T + i) * HH + jj];
    }
  }
  outv[(size_t)(n * DH + d) * T + i] += s;
}

// ---- y = fc_w (256x256) @ out (256x576) + fc_b + residual -> d_out ----------
__global__ void k_fc(const float* __restrict__ outv, const float* __restrict__ w,
                     const float* __restrict__ bias, const float* __restrict__ x,
                     float* __restrict__ y, int b)
{
  const int o0 = blockIdx.x << 4;
  const int t0 = blockIdx.y << 4;
  const int lane = threadIdx.x;
  const int m = lane & 15, hi = lane >> 4;
  v8f acc = {};
  for (int kb = 0; kb < C; kb += 32) {
    if (kb + 32 < C) __builtin_prefetch(&w[(o0 + m) * C + kb + 32], 0, 1);
    v16h av, bv;
#pragma unroll
    for (int e = 0; e < 16; ++e) {
      av[e] = (_Float16)w[(o0 + m) * C + kb + a_kidx(e, hi)];
      bv[e] = (_Float16)outv[(size_t)(kb + b_kidx(e, hi)) * T + t0 + m];
    }
    acc = __builtin_amdgcn_wmma_f32_16x16x32_f16(false, av, false, bv,
                                                 (short)0, acc, false, false);
  }
  const size_t base = (size_t)b * C * T;
#pragma unroll
  for (int r = 0; r < 8; ++r) {
    int o = o0 + r + (hi << 3);
    size_t oi = base + (size_t)o * T + t0 + m;
    y[oi] = acc[r] + bias[o] + x[oi];
  }
}

// ---- GroupNorm(16) two-pass, in place on d_out ------------------------------
__global__ void k_gnstats(const float* __restrict__ y, float* __restrict__ stats)
{
  const int bg = blockIdx.x;            // b*16 + g
  const int b = bg >> 4, g = bg & 15;
  const size_t base = (size_t)b * C * T + (size_t)g * 16 * T;
  const int N = 16 * T;                 // 9216 elems per (b,g)
  float s = 0.f, s2 = 0.f;
  for (int k = threadIdx.x; k < N; k += blockDim.x) {
    float v = y[base + k];
    s += v; s2 += v * v;
  }
  __shared__ float sh[256], sh2[256];
  sh[threadIdx.x] = s; sh2[threadIdx.x] = s2;
  __syncthreads();
  for (int off = 128; off > 0; off >>= 1) {
    if ((int)threadIdx.x < off) {
      sh[threadIdx.x]  += sh[threadIdx.x + off];
      sh2[threadIdx.x] += sh2[threadIdx.x + off];
    }
    __syncthreads();
  }
  if (threadIdx.x == 0) {
    float mu  = sh[0] / N;
    float var = sh2[0] / N - mu * mu;   // population variance (ddof=0)
    stats[2 * bg]     = mu;
    stats[2 * bg + 1] = rsqrtf(var + 1e-5f);
  }
}

__global__ void k_gnapply(float* __restrict__ y, const float* __restrict__ stats,
                          const float* __restrict__ gw, const float* __restrict__ gb)
{
  size_t idx = (size_t)blockIdx.x * blockDim.x + threadIdx.x;
  if (idx >= (size_t)NB * C * T) return;
  int c = (int)((idx / T) % C);
  int b = (int)(idx / ((size_t)C * T));
  int g = c >> 4;
  float mu = stats[2 * (b * 16 + g)];
  float rs = stats[2 * (b * 16 + g) + 1];
  y[idx] = (y[idx] - mu) * rs * gw[c] + gb[c];
}

// ---------------------------------------------------------------------------
extern "C" void kernel_launch(void* const* d_in, const int* in_sizes, int n_in,
                              void* d_out, int out_size, void* d_ws, size_t ws_size,
                              hipStream_t stream)
{
  const float* x     = (const float*)d_in[0];
  const float* qkv_w = (const float*)d_in[1];
  const float* qkv_b = (const float*)d_in[2];
  const float* rh_k  = (const float*)d_in[3];
  const float* rw_k  = (const float*)d_in[4];
  const float* rh_v  = (const float*)d_in[5];
  const float* rw_v  = (const float*)d_in[6];
  const float* fc_w  = (const float*)d_in[7];
  const float* fc_b  = (const float*)d_in[8];
  const float* gn_w  = (const float*)d_in[9];
  const float* gn_b  = (const float*)d_in[10];
  float* out = (float*)d_out;

  // Workspace (floats), reused across batches via stream ordering (~14.5 MB)
  float* ws    = (float*)d_ws;
  float* qkv   = ws;                 size_t off = (size_t)O3 * T;
  float* attn  = ws + off;           off += (size_t)NH * T * T;
  float* bdh   = ws + off;           off += (size_t)NH * T * LREL;
  float* bdw   = ws + off;           off += (size_t)NH * T * LREL;
  float* outv  = ws + off;           off += (size_t)NH * DH * T;
  float* a24   = ws + off;           off += (size_t)NH * T * HH;
  float* stats = ws + off;           off += 2 * NB * 16;

  for (int b = 0; b < NB; ++b) {
    k_qkv    <<<dim3(O3 / 16, T / 16), 32, 0, stream>>>(x, qkv_w, qkv_b, qkv, b);
    k_bd     <<<(NH * T * LREL + 255) / 256, 256, 0, stream>>>(qkv, rh_k, rw_k, bdh, bdw);
    k_ac     <<<dim3(NH, T / 16, T / 16), 32, 0, stream>>>(qkv, attn);
    k_softmax<<<(T * T + 255) / 256, 256, 0, stream>>>(attn, bdh, bdw);
    k_av     <<<dim3(NH, T / 16, DH / 16), 32, 0, stream>>>(attn, qkv, outv);
    k_a24    <<<(NH * T * HH + 255) / 256, 256, 0, stream>>>(attn, a24);
    k_pos    <<<(NH * DH * T + 255) / 256, 256, 0, stream>>>(a24, rh_v, rw_v, outv);
    k_fc     <<<dim3(C / 16, T / 16), 32, 0, stream>>>(outv, fc_w, fc_b, x, out, b);
  }
  k_gnstats<<<NB * 16, 256, 0, stream>>>(out, stats);
  k_gnapply<<<(unsigned)(((size_t)NB * C * T + 255) / 256), 256, 0, stream>>>(out, stats, gn_w, gn_b);

  (void)in_sizes; (void)n_in; (void)out_size; (void)ws_size;
}
